// MoE_56384330662294
// MI455X (gfx1250) — matrix operand order repaired and checked by reading, compiled, not run on hardware
//
#include <hip/hip_runtime.h>
#include <math.h>

// Problem constants (from the reference)
#define TOKS 4096     // B*N
#define CDIM 1024
#define HDIM 4096
#define NEXP 8
// K (top-k) = 2, TAU = 1.0

// Try the CDNA5 async global->LDS copy path (ASYNCcnt) for A-staging.
#define USE_ASYNC_LDS 1

typedef __bf16 bf16;
typedef __attribute__((ext_vector_type(8)))  float v8f;
typedef __attribute__((ext_vector_type(8)))  bf16  v8bf;
typedef __attribute__((ext_vector_type(16))) bf16  v16bf;

#define LDS_STRIDE 40   // padded bf16 stride -> conflict-free b128/b64 access

#if USE_ASYNC_LDS
// VGLOBAL async encoding: VDST = LDS byte address (VGPR), VADDR = 64-bit global addr.
#define ASYNC_COPY_B128(ldsaddr, gptr) \
  asm volatile("global_load_async_to_lds_b128 %0, %1, off" :: "v"(ldsaddr), "v"(gptr) : "memory")
#define WAIT_ASYNC() asm volatile("s_wait_asynccnt 0x0" ::: "memory")
#else
#define WAIT_ASYNC()
#endif

#define F4C(v, j) ((j) == 0 ? (v).x : (j) == 1 ? (v).y : (j) == 2 ? (v).z : (v).w)

// ---------------- workspace layout (bytes) ----------------
static constexpr size_t OFF_COUNTS = 0;                                   // int[8]
static constexpr size_t OFF_BASE   = 64;                                  // int[8]
static constexpr size_t OFF_GSUM   = 128;                                 // float[8]
static constexpr size_t OFF_TOK    = 256;                                 // int[E*T]
static constexpr size_t OFF_GATE   = OFF_TOK  + (size_t)NEXP*TOKS*4;      // float[E*T]
static constexpr size_t OFF_XB     = OFF_GATE + (size_t)NEXP*TOKS*4;      // bf16[T*C]
static constexpr size_t OFF_HBUF   = OFF_XB   + (size_t)TOKS*CDIM*2;      // bf16[T*K*H]

// ---------------- kernel 1: zero output + counters ----------------
__global__ void zero_init(float* __restrict__ out, int outN,
                          int* __restrict__ counts, float* __restrict__ gsum) {
  int tid = blockIdx.x * blockDim.x + threadIdx.x;
  int stride = gridDim.x * blockDim.x;
  float4 z = make_float4(0.f, 0.f, 0.f, 0.f);
  float4* o4 = (float4*)out;
  for (int i = tid; i < outN / 4; i += stride) o4[i] = z;
  if (tid < NEXP) { counts[tid] = 0; gsum[tid] = 0.f; }
}

// ---------------- kernel 2: x fp32 -> bf16 ----------------
__global__ void xconv(const float* __restrict__ x, bf16* __restrict__ xb) {
  int i = (blockIdx.x * blockDim.x + threadIdx.x) * 4;
  float4 v = *(const float4*)(x + i);
  union { bf16 h[4]; uint2 u; } pk;
  pk.h[0] = (bf16)v.x; pk.h[1] = (bf16)v.y; pk.h[2] = (bf16)v.z; pk.h[3] = (bf16)v.w;
  *(uint2*)(xb + i) = pk.u;
}

// ---------------- kernel 3: gate / router ----------------
__global__ __launch_bounds__(256)
void gate_kernel(const float* __restrict__ x, const float* __restrict__ gnoise,
                 const float* __restrict__ gw, const float* __restrict__ gb,
                 int* __restrict__ counts, float* __restrict__ gsum,
                 int* __restrict__ toklist, float* __restrict__ gatelist) {
  int wave = threadIdx.x >> 5;
  int lane = threadIdx.x & 31;
  int t = blockIdx.x * (blockDim.x >> 5) + wave;
  if (t >= TOKS) return;

  float acc[NEXP];
#pragma unroll
  for (int e = 0; e < NEXP; e++) acc[e] = 0.f;

  const float* xr = x + (size_t)t * CDIM;
  for (int c = lane; c < CDIM; c += 32) {
    float xv = xr[c];
    const float4* g = (const float4*)(gw + (size_t)c * NEXP);
    float4 g0 = g[0], g1 = g[1];
    acc[0] += xv * g0.x; acc[1] += xv * g0.y; acc[2] += xv * g0.z; acc[3] += xv * g0.w;
    acc[4] += xv * g1.x; acc[5] += xv * g1.y; acc[6] += xv * g1.z; acc[7] += xv * g1.w;
  }
#pragma unroll
  for (int off = 16; off >= 1; off >>= 1) {
#pragma unroll
    for (int e = 0; e < NEXP; e++) acc[e] += __shfl_down(acc[e], off, 32);
  }
  if (lane == 0) {
    float l[NEXP], m = -3.0e38f;
#pragma unroll
    for (int e = 0; e < NEXP; e++) {
      l[e] = acc[e] + gb[e] + gnoise[(size_t)t * NEXP + e];  // tau = 1.0
      m = fmaxf(m, l[e]);
    }
    float p[NEXP], s = 0.f;
#pragma unroll
    for (int e = 0; e < NEXP; e++) { p[e] = expf(l[e] - m); s += p[e]; }
    float inv = 1.f / s;
#pragma unroll
    for (int e = 0; e < NEXP; e++) { p[e] *= inv; atomicAdd(&gsum[e], p[e]); }
    int i1 = 0;
#pragma unroll
    for (int e = 1; e < NEXP; e++) if (p[e] > p[i1]) i1 = e;
    int i2 = (i1 == 0) ? 1 : 0;
#pragma unroll
    for (int e = 0; e < NEXP; e++) if (e != i1 && p[e] > p[i2]) i2 = e;
    int s1 = atomicAdd(&counts[i1], 1);
    toklist[i1 * TOKS + s1] = t; gatelist[i1 * TOKS + s1] = p[i1];
    int s2 = atomicAdd(&counts[i2], 1);
    toklist[i2 * TOKS + s2] = t; gatelist[i2 * TOKS + s2] = p[i2];
  }
}

// ---------------- kernel 4: prefix + load-balance loss ----------------
__global__ void prefix_loss(const int* __restrict__ counts, int* __restrict__ base,
                            const float* __restrict__ gsum, float* __restrict__ loss_out) {
  if (threadIdx.x == 0 && blockIdx.x == 0) {
    int acc = 0;
#pragma unroll
    for (int e = 0; e < NEXP; e++) { base[e] = acc; acc += counts[e]; }
    float L = 0.f;
#pragma unroll
    for (int e = 0; e < NEXP; e++) {
      float mg = gsum[e] / (float)TOKS;
      L += mg * logf(mg + 1e-8f);
    }
    *loss_out = L;
  }
}

// ---------------- WMMA compute phase (shared by both GEMMs) ----------------
// Block 256 = 8 wave32. Tile M=128, N=128, Kstep=32, double-buffered LDS.
__device__ __forceinline__ void wmma_compute(const bf16* __restrict__ As,
                                             const bf16* __restrict__ Bs,
                                             int mrow, int laneN, int aK0, int bK0,
                                             v8f acc[8]) {
  v8bf alo = *(const v8bf*)&As[mrow * LDS_STRIDE + aK0];
  v8bf ahi = *(const v8bf*)&As[mrow * LDS_STRIDE + aK0 + 16];
  v16bf afrag = __builtin_shufflevector(alo, ahi, 0,1,2,3,4,5,6,7,8,9,10,11,12,13,14,15);
#pragma unroll
  for (int j = 0; j < 8; j++) {
    int n = j * 16 + laneN;
    v8bf blo = *(const v8bf*)&Bs[n * LDS_STRIDE + bK0];
    v8bf bhi = *(const v8bf*)&Bs[n * LDS_STRIDE + bK0 + 16];
    v16bf bfrag = __builtin_shufflevector(blo, bhi, 0,1,2,3,4,5,6,7,8,9,10,11,12,13,14,15);
    acc[j] = __builtin_amdgcn_wmma_f32_16x16x32_bf16(
        false, afrag, false, bfrag, (short)0, acc[j], false, false);
  }
}

// kernel 5: h = gelu(gather(xb) @ w1[e] + b1[e]) -> bf16 hbuf (compact rows)
__global__ __launch_bounds__(256)
void gemm1(const bf16* __restrict__ xb, const float* __restrict__ w1,
           const float* __restrict__ b1,
           const int* __restrict__ counts, const int* __restrict__ base,
           const int* __restrict__ toklist, bf16* __restrict__ hbuf) {
  const int e   = blockIdx.z;
  const int cnt = counts[e];
  const int m0  = blockIdx.y * 128;
  if (m0 >= cnt) return;
  const int n0  = blockIdx.x * 128;

  __shared__ bf16 As[2][128 * LDS_STRIDE];
  __shared__ bf16 Bs[2][128 * LDS_STRIDE];

  const int tid = threadIdx.x, wv = tid >> 5, ln = tid & 31;
  const int laneN = ln & 15;
  const int aK0 = (ln < 16) ? 0 : 8;    // A frag: lanes 0-15 K{0..7,16..23}, 16-31 K{8..15,24..31}
  const int bK0 = (ln < 16) ? 0 : 8;    // B frag: lanes 0-15 K{0..7,16..23}, 16-31 K{8..15,24..31}
  const int mrow = 16 * wv + laneN;

  // A staging: thread -> (row tid/2, 16 cols at (tid&1)*16); clamp invalid rows
  const int ar = tid >> 1, ah = (tid & 1) * 16;
  const int arow = (m0 + ar < cnt) ? (m0 + ar) : (cnt - 1);
  const bf16* asrc = xb + (size_t)toklist[e * TOKS + arow] * CDIM + ah;

  // B staging: thread -> (K-quad kq, N-quad nq); 4 float4 loads, 4 ds_store_b64
  const int kq = tid >> 5;              // 0..7  -> k rows 4*kq..4*kq+3
  const int nq = tid & 31;              // 0..31 -> n cols 4*nq..4*nq+3
  const float* wE = w1 + (size_t)e * CDIM * HDIM;

  v8f acc[8];
  v8f zero = {};
#pragma unroll
  for (int j = 0; j < 8; j++) acc[j] = zero;

  float4 bR[4];
  const int NK = CDIM / 32;

#define G1_STAGE_A(buf, kk)                                                   \
  {                                                                           \
    unsigned l0 = (unsigned)(size_t)&As[buf][ar * LDS_STRIDE + ah];           \
    ASYNC_COPY_B128(l0, asrc + (kk));                                         \
    ASYNC_COPY_B128(l0 + 16u, asrc + (kk) + 8);                               \
  }
#define G1_LOAD_B(kk)                                                         \
  {                                                                           \
    _Pragma("unroll") for (int dk = 0; dk < 4; dk++)                          \
      bR[dk] = *(const float4*)(wE + (size_t)((kk) + kq * 4 + dk) * HDIM + n0 + nq * 4); \
  }
#define G1_STORE_B(buf)                                                       \
  {                                                                           \
    _Pragma("unroll") for (int j = 0; j < 4; j++) {                           \
      union { bf16 h[4]; uint2 u; } pk;                                       \
      _Pragma("unroll") for (int dk = 0; dk < 4; dk++) pk.h[dk] = (bf16)F4C(bR[dk], j); \
      *(uint2*)&Bs[buf][(nq * 4 + j) * LDS_STRIDE + kq * 4] = pk.u;           \
    }                                                                         \
  }

#if !USE_ASYNC_LDS
  uint4 aR0, aR1;
#undef G1_STAGE_A
#define G1_STAGE_A(buf, kk)                                                   \
  {                                                                           \
    aR0 = ((const uint4*)(asrc + (kk)))[0];                                   \
    aR1 = ((const uint4*)(asrc + (kk)))[1];                                   \
    *(uint4*)&As[buf][ar * LDS_STRIDE + ah]     = aR0;                        \
    *(uint4*)&As[buf][ar * LDS_STRIDE + ah + 8] = aR1;                        \
  }
#endif

  // prologue: stage tile 0 into buffer 0
  G1_STAGE_A(0, 0);
  G1_LOAD_B(0);
  G1_STORE_B(0);
  WAIT_ASYNC();
  __syncthreads();

  for (int kt = 0; kt < NK; kt++) {
    const int c = kt & 1, nb = c ^ 1;
    const bool more = (kt + 1) < NK;
    const int kkn = (kt + 1) * 32;
    if (more) {
      G1_STAGE_A(nb, kkn);
      __builtin_prefetch(wE + (size_t)(kkn + 32) * HDIM + n0, 0, 1);
      G1_LOAD_B(kkn);
    }
    wmma_compute(As[c], Bs[c], mrow, laneN, aK0, bK0, acc);
    if (more) { G1_STORE_B(nb); }
    WAIT_ASYNC();
    __syncthreads();
  }

  // epilogue: +bias, exact GELU, -> bf16 hbuf
  const int mBase = m0 + 16 * wv + ((ln < 16) ? 0 : 8);
  const int hrow0 = base[e];
#pragma unroll
  for (int j = 0; j < 8; j++) {
    int n = n0 + j * 16 + laneN;
    float bias = b1[e * HDIM + n];
#pragma unroll
    for (int r = 0; r < 8; r++) {
      int m = mBase + r;
      if (m < cnt) {
        float v = acc[j][r] + bias;
        v = 0.5f * v * (1.0f + erff(v * 0.70710678118654752f));
        hbuf[(size_t)(hrow0 + m) * HDIM + n] = (bf16)v;
      }
    }
  }
}

// kernel 6: y = h @ w2[e] + b2[e]; out[tok,:] += gate * y (atomic scatter)
__global__ __launch_bounds__(256)
void gemm2(const bf16* __restrict__ hbuf, const float* __restrict__ w2,
           const float* __restrict__ b2,
           const int* __restrict__ counts, const int* __restrict__ base,
           const int* __restrict__ toklist, const float* __restrict__ gatelist,
           float* __restrict__ out) {
  const int e   = blockIdx.z;
  const int cnt = counts[e];
  const int m0  = blockIdx.y * 128;
  if (m0 >= cnt) return;
  const int n0  = blockIdx.x * 128;

  __shared__ bf16 As[2][128 * LDS_STRIDE];
  __shared__ bf16 Bs[2][128 * LDS_STRIDE];

  const int tid = threadIdx.x, wv = tid >> 5, ln = tid & 31;
  const int laneN = ln & 15;
  const int aK0 = (ln < 16) ? 0 : 8;
  const int bK0 = (ln < 16) ? 0 : 8;
  const int mrow = 16 * wv + laneN;

  const int ar = tid >> 1, ah = (tid & 1) * 16;
  const int arow = (m0 + ar < cnt) ? (m0 + ar) : (cnt - 1);
  const bf16* asrc = hbuf + (size_t)(base[e] + arow) * HDIM + ah;

  const int kq = tid >> 5;
  const int nq = tid & 31;
  const float* wE = w2 + (size_t)e * HDIM * CDIM;

  v8f acc[8];
  v8f zero = {};
#pragma unroll
  for (int j = 0; j < 8; j++) acc[j] = zero;

  float4 bR[4];
  const int NK = HDIM / 32;

#define G2_STAGE_A(buf, kk)                                                   \
  {                                                                           \
    unsigned l0 = (unsigned)(size_t)&As[buf][ar * LDS_STRIDE + ah];           \
    ASYNC_COPY_B128(l0, asrc + (kk));                                         \
    ASYNC_COPY_B128(l0 + 16u, asrc + (kk) + 8);                               \
  }
#define G2_LOAD_B(kk)                                                         \
  {                                                                           \
    _Pragma("unroll") for (int dk = 0; dk < 4; dk++)                          \
      bR[dk] = *(const float4*)(wE + (size_t)((kk) + kq * 4 + dk) * CDIM + n0 + nq * 4); \
  }
#define G2_STORE_B(buf)                                                       \
  {                                                                           \
    _Pragma("unroll") for (int j = 0; j < 4; j++) {                           \
      union { bf16 h[4]; uint2 u; } pk;                                       \
      _Pragma("unroll") for (int dk = 0; dk < 4; dk++) pk.h[dk] = (bf16)F4C(bR[dk], j); \
      *(uint2*)&Bs[buf][(nq * 4 + j) * LDS_STRIDE + kq * 4] = pk.u;           \
    }                                                                         \
  }

#if !USE_ASYNC_LDS
  uint4 aR0, aR1;
#undef G2_STAGE_A
#define G2_STAGE_A(buf, kk)                                                   \
  {                                                                           \
    aR0 = ((const uint4*)(asrc + (kk)))[0];                                   \
    aR1 = ((const uint4*)(asrc + (kk)))[1];                                   \
    *(uint4*)&As[buf][ar * LDS_STRIDE + ah]     = aR0;                        \
    *(uint4*)&As[buf][ar * LDS_STRIDE + ah + 8] = aR1;                        \
  }
#endif

  G2_STAGE_A(0, 0);
  G2_LOAD_B(0);
  G2_STORE_B(0);
  WAIT_ASYNC();
  __syncthreads();

  for (int kt = 0; kt < NK; kt++) {
    const int c = kt & 1, nb = c ^ 1;
    const bool more = (kt + 1) < NK;
    const int kkn = (kt + 1) * 32;
    if (more) {
      G2_STAGE_A(nb, kkn);
      __builtin_prefetch(wE + (size_t)(kkn + 32) * CDIM + n0, 0, 1);
      G2_LOAD_B(kkn);
    }
    wmma_compute(As[c], Bs[c], mrow, laneN, aK0, bK0, acc);
    if (more) { G2_STORE_B(nb); }
    WAIT_ASYNC();
    __syncthreads();
  }

  // epilogue: per-row token + gate, atomic scatter-add into out
  const int mBase = m0 + 16 * wv + ((ln < 16) ? 0 : 8);
  int   tk[8];
  float gt[8];
#pragma unroll
  for (int r = 0; r < 8; r++) {
    int m = mBase + r;
    if (m < cnt) {
      tk[r] = toklist[e * TOKS + m];
      gt[r] = gatelist[e * TOKS + m];
    } else {
      tk[r] = -1; gt[r] = 0.f;
    }
  }
#pragma unroll
  for (int j = 0; j < 8; j++) {
    int n = n0 + j * 16 + laneN;
    float bias = b2[e * CDIM + n];
#pragma unroll
    for (int r = 0; r < 8; r++) {
      if (tk[r] >= 0) {
        float y = acc[j][r] + bias;
        atomicAdd(&out[(size_t)tk[r] * CDIM + n], gt[r] * y);
      }
    }
  }
}

// ---------------- launcher ----------------
extern "C" void kernel_launch(void* const* d_in, const int* in_sizes, int n_in,
                              void* d_out, int out_size, void* d_ws, size_t ws_size,
                              hipStream_t stream) {
  const float* x  = (const float*)d_in[0];
  const float* gn = (const float*)d_in[1];
  const float* gw = (const float*)d_in[2];
  const float* gb = (const float*)d_in[3];
  const float* w1 = (const float*)d_in[4];
  const float* b1 = (const float*)d_in[5];
  const float* w2 = (const float*)d_in[6];
  const float* b2 = (const float*)d_in[7];
  float* out = (float*)d_out;

  char* ws = (char*)d_ws;
  int*   counts   = (int*)  (ws + OFF_COUNTS);
  int*   basep    = (int*)  (ws + OFF_BASE);
  float* gsum     = (float*)(ws + OFF_GSUM);
  int*   toklist  = (int*)  (ws + OFF_TOK);
  float* gatelist = (float*)(ws + OFF_GATE);
  bf16*  xb       = (bf16*) (ws + OFF_XB);
  bf16*  hbuf     = (bf16*) (ws + OFF_HBUF);

  const int outN = TOKS * CDIM;  // loss scalar lives at out[outN]

  zero_init<<<1024, 256, 0, stream>>>(out, outN, counts, gsum);
  xconv<<<(TOKS * CDIM / 4) / 256, 256, 0, stream>>>(x, xb);
  gate_kernel<<<TOKS / 8, 256, 0, stream>>>(x, gn, gw, gb, counts, gsum, toklist, gatelist);
  prefix_loss<<<1, 32, 0, stream>>>(counts, basep, gsum, out + outN);
  gemm1<<<dim3(HDIM / 128, TOKS / 128, NEXP), 256, 0, stream>>>(
      xb, w1, b1, counts, basep, toklist, hbuf);
  gemm2<<<dim3(CDIM / 128, TOKS / 128, NEXP), 256, 0, stream>>>(
      hbuf, w2, b2, counts, basep, toklist, gatelist, out);
}
